// PaiNN_2619930050849
// MI455X (gfx1250) — matrix operand order repaired and checked by reading, not compiled, not run on hardware
//
#include <hip/hip_runtime.h>
#include <hip/hip_bf16.h>
#include <stdint.h>

#define FDIM   128
#define RBFN   20
#define RPAD   32
#define C3F    384
#define NLAYER 3
#define CUTOFF 3.0f
#define PI_F   3.14159265358979f

typedef __attribute__((ext_vector_type(16))) __bf16 v16bf;
typedef __attribute__((ext_vector_type(8)))  float  v8f;

__device__ __forceinline__ unsigned short f2bf(float x) {
    unsigned int u = __float_as_uint(x);
    unsigned int r = (u + 0x7FFFu + ((u >> 16) & 1u)) >> 16;
    return (unsigned short)r;
}

union FragBF { uint4 q[2]; v16bf v; };

// Load a 16-element bf16 fragment from two 16B chunks (works for global or LDS)
__device__ __forceinline__ v16bf load_frag2(const void* p0, const void* p1) {
    FragBF f;
    f.q[0] = *(const uint4*)p0;
    f.q[1] = *(const uint4*)p1;
    return f.v;
}

// ---------------------------------------------------------------------------
// Asynchronously copy `bytes` (multiple of 16) from global to LDS using the
// CDNA5 async-DMA path (GLOBAL_LOAD_ASYNC_TO_LDS_B128, tracked by ASYNCcnt),
// then wait on this wave's ASYNCcnt. Caller must barrier afterwards.
// ---------------------------------------------------------------------------
__device__ __forceinline__ void async_stage(const void* gsrc, void* lds,
                                            int bytes, int tid, int nthr) {
    unsigned lbase = (unsigned)(size_t)lds;   // wave-relative LDS byte offset
    const char* g = (const char*)gsrc;
    for (int it = tid; it < bytes / 16; it += nthr) {
        unsigned loff = lbase + it * 16;
        const char* ga = g + (size_t)it * 16;
        asm volatile("global_load_async_to_lds_b128 %0, %1, off"
                     :: "v"(loff), "v"(ga) : "memory");
    }
    asm volatile("s_wait_asynccnt 0x0" ::: "memory");
}

// ---------------------------------------------------------------------------
// Weight packing: f32 [K, COLS] row-major -> bf16 in WMMA B-fragment order.
// dword index t = ((ct*Ksteps + ks)*32 + lane)*8 + v
//   k = ks*32 + (lane>>4)*8 + (v>>2)*16 + 2*(v&3), n = ct*16 + (lane&15)
// ---------------------------------------------------------------------------
__global__ void pack_w(const float* __restrict__ src, unsigned int* __restrict__ dst,
                       int realK, int K, int COLS) {
    int t = blockIdx.x * blockDim.x + threadIdx.x;
    int Ksteps = K / 32;
    int total = (COLS / 16) * Ksteps * 32 * 8;
    if (t >= total) return;
    int v    = t & 7;
    int lane = (t >> 3) & 31;
    int rest = t >> 8;
    int ks   = rest % Ksteps;
    int ct   = rest / Ksteps;
    int k = ks * 32 + (lane >> 4) * 8 + (v >> 2) * 16 + 2 * (v & 3);
    int n = ct * 16 + (lane & 15);
    float a = (k     < realK) ? src[(size_t)k       * COLS + n] : 0.f;
    float b = (k + 1 < realK) ? src[(size_t)(k + 1) * COLS + n] : 0.f;
    dst[t] = (unsigned int)f2bf(a) | ((unsigned int)f2bf(b) << 16);
}

// ---------------------------------------------------------------------------
// Generic WMMA GEMM: out = act(A[rows,K]_bf16 @ Wpacked[K,COLS] + bias)
// 8 waves / block, 16 rows / wave. Whole packed W staged async into LDS once
// per block; B fragments then come from LDS (ds_load_b128), shared by 8 waves.
// ACT: 0 = none, 1 = SiLU. outF (f32) and/or outBF (bf16) written if non-null.
// ---------------------------------------------------------------------------
template <int K, int COLS, int ACT>
__global__ __launch_bounds__(256) void gemm_wmma(
    const unsigned short* __restrict__ A, const unsigned int* __restrict__ Wp,
    const float* __restrict__ bias, float* __restrict__ outF,
    unsigned short* __restrict__ outBF, int rows) {
    constexpr int KS = K / 32;
    constexpr int CT = COLS / 16;
    constexpr int WBYTES = K * COLS * 2;          // packed bf16 weight bytes
    __shared__ __align__(16) unsigned int ldsW[WBYTES / 4];

    async_stage(Wp, ldsW, WBYTES, threadIdx.x, 256);
    __syncthreads();

    int wave = threadIdx.x >> 5;
    int lane = threadIdx.x & 31;
    int tile = blockIdx.x * 8 + wave;
    if (tile * 16 >= rows) return;
    int row0 = tile * 16;
    int half = lane >> 4;
    int m    = lane & 15;

    // A fragments for all K-steps live in registers (16xK bf16 tile / wave)
    const char* abase = (const char*)A + ((size_t)(row0 + m) * K + (size_t)half * 8) * 2;
    v16bf afrag[KS];
#pragma unroll
    for (int ks = 0; ks < KS; ++ks)
        afrag[ks] = load_frag2(abase + ks * 64, abase + ks * 64 + 32);

    for (int ct = 0; ct < CT; ++ct) {
        const char* wbase = (const char*)ldsW + (((size_t)ct * KS) * 32 + lane) * 32;
        v8f acc = {0.f, 0.f, 0.f, 0.f, 0.f, 0.f, 0.f, 0.f};
#pragma unroll
        for (int ks = 0; ks < KS; ++ks) {
            v16bf bfrag = load_frag2(wbase + (size_t)ks * 1024,
                                     wbase + (size_t)ks * 1024 + 16);
            acc = __builtin_amdgcn_wmma_f32_16x16x32_bf16(
                false, afrag[ks], false, bfrag, (short)0, acc, false, false);
        }
        int c = ct * 16 + m;
        float bz = bias ? bias[c] : 0.f;
#pragma unroll
        for (int vg = 0; vg < 8; ++vg) {
            float x = acc[vg] + bz;
            if (ACT == 1) x = x / (1.f + __expf(-x));
            int row = row0 + vg + 8 * half;
            if (outF)  outF [(size_t)row * COLS + c] = x;
            if (outBF) outBF[(size_t)row * COLS + c] = f2bf(x);
        }
    }
}

// ---------------------------------------------------------------------------
// Edge message kernel. One wave handles a 16-edge tile; 4 waves / block.
// Packed rbf_w (24KB) staged async into LDS once per block. The 24 col tiles
// are split into three specialized loops (vv / ss / vs) - no divergent part
// selection in the hot loop.
// ---------------------------------------------------------------------------
__global__ __launch_bounds__(128) void edge_msg(
    const unsigned short* __restrict__ rbf_bf, const unsigned int* __restrict__ Wp,
    const float* __restrict__ bias, const float* __restrict__ cutw,
    const float* __restrict__ phi, const int* __restrict__ idx_i,
    const int* __restrict__ idx_j, const float* __restrict__ dir,
    const float* __restrict__ vold, float* __restrict__ vnew,
    float* __restrict__ sfeat, int En, int Nn) {
    constexpr int WBYTES = RPAD * C3F * 2;        // 24576 bytes
    __shared__ __align__(16) unsigned int ldsW[WBYTES / 4];

    async_stage(Wp, ldsW, WBYTES, threadIdx.x, 128);
    __syncthreads();

    int wave = threadIdx.x >> 5;
    int lane = threadIdx.x & 31;
    int e0 = (blockIdx.x * 4 + wave) * 16;
    if (e0 >= En) return;
    int half = lane >> 4;
    int m    = lane & 15;

    int arow = e0 + m; if (arow >= En) arow = En - 1;
    const char* abase = (const char*)rbf_bf + (size_t)arow * (RPAD * 2) + (size_t)half * 16;
    v16bf afrag = load_frag2(abase, abase + 32);

    int   ei[8], ej[8];
    float ec[8], dv0[8], dv1[8], dv2[8];
#pragma unroll
    for (int vg = 0; vg < 8; ++vg) {
        int e = e0 + vg + 8 * half;
        int ecl = (e < En) ? e : (En - 1);
        ei[vg] = idx_i[ecl];
        ej[vg] = idx_j[ecl];
        ec[vg] = (e < En) ? cutw[ecl] : 0.f;
        dv0[vg] = dir[(size_t)ecl * 3 + 0];
        dv1[vg] = dir[(size_t)ecl * 3 + 1];
        dv2[vg] = dir[(size_t)ecl * 3 + 2];
    }

    // ---- part 0 (vv): cols 0..127 -> vnew[i] += vold[j] * phiW ----
    for (int ct = 0; ct < 8; ++ct) {
        const char* wbase = (const char*)ldsW + ((size_t)ct * 32 + lane) * 32;
        v16bf bfrag = load_frag2(wbase, wbase + 16);
        v8f acc = {0.f, 0.f, 0.f, 0.f, 0.f, 0.f, 0.f, 0.f};
        acc = __builtin_amdgcn_wmma_f32_16x16x32_bf16(
            false, afrag, false, bfrag, (short)0, acc, false, false);
        int c = ct * 16 + m;                   // also the feature index f
        float bz = bias[c];
#pragma unroll
        for (int vg = 0; vg < 8; ++vg) {
            float w  = (acc[vg] + bz) * ec[vg];
            int   j  = ej[vg], i = ei[vg];
            float pw = phi[(size_t)j * C3F + c] * w;
#pragma unroll
            for (int d = 0; d < 3; ++d) {
                float val = vold[((size_t)d * Nn + j) * FDIM + c] * pw;
                atomicAdd(&vnew[((size_t)d * Nn + i) * FDIM + c], val);
            }
        }
    }
    // ---- part 1 (ss): cols 128..255 -> sfeat[i] += phiW ----
    for (int ct = 8; ct < 16; ++ct) {
        const char* wbase = (const char*)ldsW + ((size_t)ct * 32 + lane) * 32;
        v16bf bfrag = load_frag2(wbase, wbase + 16);
        v8f acc = {0.f, 0.f, 0.f, 0.f, 0.f, 0.f, 0.f, 0.f};
        acc = __builtin_amdgcn_wmma_f32_16x16x32_bf16(
            false, afrag, false, bfrag, (short)0, acc, false, false);
        int c = ct * 16 + m;
        int f = c - 128;
        float bz = bias[c];
#pragma unroll
        for (int vg = 0; vg < 8; ++vg) {
            float w  = (acc[vg] + bz) * ec[vg];
            int   j  = ej[vg], i = ei[vg];
            float pw = phi[(size_t)j * C3F + c] * w;
            atomicAdd(&sfeat[(size_t)i * FDIM + f], pw);
        }
    }
    // ---- part 2 (vs): cols 256..383 -> vnew[i] += phiW * dir ----
    for (int ct = 16; ct < 24; ++ct) {
        const char* wbase = (const char*)ldsW + ((size_t)ct * 32 + lane) * 32;
        v16bf bfrag = load_frag2(wbase, wbase + 16);
        v8f acc = {0.f, 0.f, 0.f, 0.f, 0.f, 0.f, 0.f, 0.f};
        acc = __builtin_amdgcn_wmma_f32_16x16x32_bf16(
            false, afrag, false, bfrag, (short)0, acc, false, false);
        int c = ct * 16 + m;
        int f = c - 256;
        float bz = bias[c];
#pragma unroll
        for (int vg = 0; vg < 8; ++vg) {
            float w  = (acc[vg] + bz) * ec[vg];
            int   j  = ej[vg], i = ei[vg];
            float pw = phi[(size_t)j * C3F + c] * w;
            atomicAdd(&vnew[((size_t)0 * Nn + i) * FDIM + f], pw * dv0[vg]);
            atomicAdd(&vnew[((size_t)1 * Nn + i) * FDIM + f], pw * dv1[vg]);
            atomicAdd(&vnew[((size_t)2 * Nn + i) * FDIM + f], pw * dv2[vg]);
        }
    }
}

// ---------------------------------------------------------------------------
// Elementwise kernels
// ---------------------------------------------------------------------------
__global__ void init_kernel(const int* __restrict__ z, const float* __restrict__ emb,
                            float* __restrict__ sfeat, unsigned short* __restrict__ sbf,
                            float* __restrict__ vA, int Nn) {
    size_t t = (size_t)blockIdx.x * blockDim.x + threadIdx.x;
    size_t tot = (size_t)Nn * FDIM;
    if (t < tot) {
        int n = (int)(t / FDIM), f = (int)(t % FDIM);
        float v = emb[(size_t)z[n] * FDIM + f];
        sfeat[t] = v;
        sbf[t]   = f2bf(v);
    }
    if (t < 3 * tot) vA[t] = 0.f;
}

__global__ void geom_kernel(const float* __restrict__ pos, const int* __restrict__ idx_i,
                            const int* __restrict__ idx_j, unsigned short* __restrict__ rbf_bf,
                            float* __restrict__ dir, float* __restrict__ cutw, int En) {
    int e = blockIdx.x * blockDim.x + threadIdx.x;
    if (e >= En) return;
    int i = idx_i[e], j = idx_j[e];
    float dx = pos[(size_t)j * 3 + 0] - pos[(size_t)i * 3 + 0];
    float dy = pos[(size_t)j * 3 + 1] - pos[(size_t)i * 3 + 1];
    float dz = pos[(size_t)j * 3 + 2] - pos[(size_t)i * 3 + 2];
    float sq = dx * dx + dy * dy + dz * dz;
    float dist = sqrtf(sq);
    float inv  = (dist > 0.f) ? (1.f / dist) : 0.f;
    dir[(size_t)e * 3 + 0] = dx * inv;
    dir[(size_t)e * 3 + 1] = dy * inv;
    dir[(size_t)e * 3 + 2] = dz * inv;
    cutw[e] = (dist < CUTOFF) ? 0.5f * (__cosf(PI_F * dist / CUTOFF) + 1.f) : 0.f;
#pragma unroll
    for (int r = 0; r < RPAD; ++r) {
        float v = 0.f;
        if (r < RBFN) v = __sinf((float)(r + 1) * PI_F * dist / CUTOFF) * inv;
        rbf_bf[(size_t)e * RPAD + r] = f2bf(v);
    }
}

__global__ void refresh_k(const float* __restrict__ sfeat, const float* __restrict__ v,
                          unsigned short* __restrict__ sbf, unsigned short* __restrict__ vbf,
                          int Nn) {
    size_t t = (size_t)blockIdx.x * blockDim.x + threadIdx.x;
    size_t tot = (size_t)Nn * FDIM;
    if (t < tot) sbf[t] = f2bf(sfeat[t]);
    if (t < 3 * tot) vbf[t] = f2bf(v[t]);
}

__global__ void build_x(const float* __restrict__ Vv, const unsigned short* __restrict__ sbf,
                        unsigned short* __restrict__ xbf, int Nn) {
    size_t t = (size_t)blockIdx.x * blockDim.x + threadIdx.x;
    size_t tot = (size_t)Nn * FDIM;
    if (t >= tot) return;
    int n = (int)(t / FDIM), f = (int)(t % FDIM);
    float s = 0.f;
#pragma unroll
    for (int d = 0; d < 3; ++d) {
        float u = Vv[((size_t)d * Nn + n) * FDIM + f];
        s += u * u;
    }
    float norm = (s > 0.f) ? sqrtf(s) : 0.f;
    xbf[(size_t)n * 256 + f]        = f2bf(norm);
    xbf[(size_t)n * 256 + 128 + f]  = sbf[t];
}

__global__ void update_k(const float* __restrict__ Uv, const float* __restrict__ Vv,
                         const float* __restrict__ a, float* __restrict__ sfeat,
                         unsigned short* __restrict__ sbf, float* __restrict__ v, int Nn) {
    size_t t = (size_t)blockIdx.x * blockDim.x + threadIdx.x;
    size_t tot = (size_t)Nn * FDIM;
    if (t >= tot) return;
    int n = (int)(t / FDIM), f = (int)(t % FDIM);
    float avv = a[(size_t)n * C3F + f];
    float asv = a[(size_t)n * C3F + 128 + f];
    float ass = a[(size_t)n * C3F + 256 + f];
    float dot = 0.f;
#pragma unroll
    for (int d = 0; d < 3; ++d) {
        size_t idx = ((size_t)d * Nn + n) * FDIM + f;
        float u = Uv[idx];
        dot += u * Vv[idx];
        v[idx] += u * avv;
    }
    float s = sfeat[t] + ass + asv * dot;
    sfeat[t] = s;
    sbf[t]   = f2bf(s);
}

__global__ void final_k(const float* __restrict__ sfeat, const float* __restrict__ v,
                        float* __restrict__ out, int Nn) {
    size_t t = (size_t)blockIdx.x * blockDim.x + threadIdx.x;
    size_t tot = (size_t)Nn * FDIM;
    if (t >= tot) return;
    out[t] = sfeat[t];
    int n = (int)(t / FDIM), f = (int)(t % FDIM);
#pragma unroll
    for (int d = 0; d < 3; ++d)
        out[tot + ((size_t)n * FDIM + f) * 3 + d] = v[((size_t)d * Nn + n) * FDIM + f];
}

// ---------------------------------------------------------------------------
// Host orchestration
// ---------------------------------------------------------------------------
extern "C" void kernel_launch(void* const* d_in, const int* in_sizes, int n_in,
                              void* d_out, int out_size, void* d_ws, size_t ws_size,
                              hipStream_t stream) {
    (void)n_in; (void)out_size; (void)ws_size;
    const int*   z      = (const int*)d_in[0];
    const float* pos    = (const float*)d_in[1];
    const int*   idx_i  = (const int*)d_in[2];
    const int*   idx_j  = (const int*)d_in[3];
    const float* emb    = (const float*)d_in[4];
    const float* msg_w1 = (const float*)d_in[5];
    const float* msg_b1 = (const float*)d_in[6];
    const float* msg_w2 = (const float*)d_in[7];
    const float* msg_b2 = (const float*)d_in[8];
    const float* rbf_w  = (const float*)d_in[9];
    const float* rbf_b  = (const float*)d_in[10];
    const float* upd_U  = (const float*)d_in[11];
    const float* upd_V  = (const float*)d_in[12];
    const float* upd_w1 = (const float*)d_in[13];
    const float* upd_b1 = (const float*)d_in[14];
    const float* upd_w2 = (const float*)d_in[15];
    const float* upd_b2 = (const float*)d_in[16];

    int Nn = in_sizes[0];
    int En = in_sizes[2];
    size_t NF = (size_t)Nn * FDIM;

    char* ws = (char*)d_ws;
    size_t off = 0;
    auto alloc = [&](size_t bytes) -> char* {
        char* p = ws + off;
        off = (off + bytes + 255) & ~(size_t)255;
        return p;
    };
    float*          sfeat  = (float*)alloc(NF * 4);
    unsigned short* sbf    = (unsigned short*)alloc(NF * 2);
    float*          vA     = (float*)alloc(3 * NF * 4);
    float*          vB     = (float*)alloc(3 * NF * 4);
    unsigned short* vbf    = (unsigned short*)alloc(3 * NF * 2);
    unsigned short* hbf    = (unsigned short*)alloc(NF * 2);
    float*          phi    = (float*)alloc(3 * NF * 4);
    float*          Uv     = (float*)alloc(3 * NF * 4);
    float*          Vvb    = (float*)alloc(3 * NF * 4);
    unsigned short* xbf    = (unsigned short*)alloc(2 * NF * 2);
    unsigned short* u1bf   = (unsigned short*)alloc(NF * 2);
    float*          abuf   = (float*)alloc(3 * NF * 4);
    unsigned short* rbf_bf = (unsigned short*)alloc((size_t)En * RPAD * 2);
    float*          dirb   = (float*)alloc((size_t)En * 3 * 4);
    float*          cutw   = (float*)alloc((size_t)En * 4);

    auto packsz = [](int K, int C) { return (size_t)K * C / 2 * 4; };
    unsigned int *mw1p[NLAYER], *mw2p[NLAYER], *rwp[NLAYER],
                 *uUp[NLAYER], *uVp[NLAYER], *uw1p[NLAYER], *uw2p[NLAYER];
    for (int l = 0; l < NLAYER; ++l) {
        mw1p[l] = (unsigned int*)alloc(packsz(128, 128));
        mw2p[l] = (unsigned int*)alloc(packsz(128, 384));
        rwp[l]  = (unsigned int*)alloc(packsz(32, 384));
        uUp[l]  = (unsigned int*)alloc(packsz(128, 128));
        uVp[l]  = (unsigned int*)alloc(packsz(128, 128));
        uw1p[l] = (unsigned int*)alloc(packsz(256, 128));
        uw2p[l] = (unsigned int*)alloc(packsz(128, 384));
    }

    auto launch_pack = [&](const float* src, unsigned int* dst, int realK, int K, int C) {
        int total = (C / 16) * (K / 32) * 256;
        hipLaunchKernelGGL(pack_w, dim3((total + 255) / 256), dim3(256), 0, stream,
                           src, dst, realK, K, C);
    };
    for (int l = 0; l < NLAYER; ++l) {
        launch_pack(msg_w1 + (size_t)l * 128 * 128, mw1p[l], 128, 128, 128);
        launch_pack(msg_w2 + (size_t)l * 128 * 384, mw2p[l], 128, 128, 384);
        launch_pack(rbf_w  + (size_t)l * RBFN * 384, rwp[l], RBFN, 32, 384);
        launch_pack(upd_U  + (size_t)l * 128 * 128, uUp[l], 128, 128, 128);
        launch_pack(upd_V  + (size_t)l * 128 * 128, uVp[l], 128, 128, 128);
        launch_pack(upd_w1 + (size_t)l * 256 * 128, uw1p[l], 256, 256, 128);
        launch_pack(upd_w2 + (size_t)l * 128 * 384, uw2p[l], 128, 128, 384);
    }

    {
        size_t tot = 3 * NF;
        init_kernel<<<dim3((unsigned)((tot + 255) / 256)), dim3(256), 0, stream>>>(
            z, emb, sfeat, sbf, vA, Nn);
    }
    geom_kernel<<<dim3((En + 255) / 256), dim3(256), 0, stream>>>(
        pos, idx_i, idx_j, rbf_bf, dirb, cutw, En);

    int tilesN = Nn / 16;
    int gB = (tilesN + 7) / 8;
    unsigned gEl = (unsigned)((NF + 255) / 256);
    unsigned gEl3 = (unsigned)((3 * NF + 255) / 256);

    for (int l = 0; l < NLAYER; ++l) {
        // ---- message block ----
        gemm_wmma<128, 128, 1><<<gB, 256, 0, stream>>>(
            sbf, mw1p[l], msg_b1 + (size_t)l * 128, (float*)nullptr, hbf, Nn);
        gemm_wmma<128, 384, 0><<<gB, 256, 0, stream>>>(
            hbf, mw2p[l], msg_b2 + (size_t)l * 384, phi, (unsigned short*)nullptr, Nn);

        hipMemcpyAsync(vB, vA, 3 * NF * 4, hipMemcpyDeviceToDevice, stream);
        edge_msg<<<dim3((En + 63) / 64), dim3(128), 0, stream>>>(
            rbf_bf, rwp[l], rbf_b + (size_t)l * 384, cutw, phi, idx_i, idx_j,
            dirb, vA, vB, sfeat, En, Nn);
        { float* t = vA; vA = vB; vB = t; }

        // ---- update block ----
        refresh_k<<<gEl3, 256, 0, stream>>>(sfeat, vA, sbf, vbf, Nn);
        for (int d = 0; d < 3; ++d) {
            gemm_wmma<128, 128, 0><<<gB, 256, 0, stream>>>(
                vbf + (size_t)d * NF, uUp[l], (const float*)nullptr,
                Uv + (size_t)d * NF, (unsigned short*)nullptr, Nn);
            gemm_wmma<128, 128, 0><<<gB, 256, 0, stream>>>(
                vbf + (size_t)d * NF, uVp[l], (const float*)nullptr,
                Vvb + (size_t)d * NF, (unsigned short*)nullptr, Nn);
        }
        build_x<<<gEl, 256, 0, stream>>>(Vvb, sbf, xbf, Nn);
        gemm_wmma<256, 128, 1><<<gB, 256, 0, stream>>>(
            xbf, uw1p[l], upd_b1 + (size_t)l * 128, (float*)nullptr, u1bf, Nn);
        gemm_wmma<128, 384, 0><<<gB, 256, 0, stream>>>(
            u1bf, uw2p[l], upd_b2 + (size_t)l * 384, abuf, (unsigned short*)nullptr, Nn);
        update_k<<<gEl, 256, 0, stream>>>(Uv, Vvb, abuf, sfeat, sbf, vA, Nn);
    }

    final_k<<<gEl, 256, 0, stream>>>(sfeat, vA, (float*)d_out, Nn);
}
